// Critic_45552423141836
// MI455X (gfx1250) — compile-verified, matrix operand order checked by reading, not run on hardware
//
#include <hip/hip_runtime.h>
#include <hip/hip_bf16.h>
#include <math.h>

// ---- problem constants (from reference) ----
#define BB   64
#define NN   256
#define DD   256
#define EE   256
#define HH   8
#define DHH  32
#define MTOT (BB * NN)   // 16384

typedef __attribute__((ext_vector_type(16))) _Float16 v16h;
typedef __attribute__((ext_vector_type(8)))  _Float16 v8h;
typedef __attribute__((ext_vector_type(8)))  float    v8f;
typedef __attribute__((ext_vector_type(4)))  int      v4i;

// ---------------------------------------------------------------------------
// CDNA5 async global->LDS copy (GLOBAL_LOAD_ASYNC_TO_LDS_B128, ASYNCcnt).
// Builtin signature (from compiler diagnostic):
//   (v4i addrspace(1)* global_src, v4i addrspace(3)* lds_dst, imm offset, imm cpol)
// Guarded: falls back to a synchronous VGPR round-trip if absent.
// ---------------------------------------------------------------------------
#if __has_builtin(__builtin_amdgcn_global_load_async_to_lds_b128)
#define HAS_ASYNC_LDS 1
#else
#define HAS_ASYNC_LDS 0
#endif

typedef __attribute__((address_space(1))) v4i* gv4i_p;   // global int4*
typedef __attribute__((address_space(3))) v4i* lv4i_p;   // LDS int4*

__device__ __forceinline__ void async_copy16(_Float16* lds, const _Float16* g) {
#if HAS_ASYNC_LDS
  __builtin_amdgcn_global_load_async_to_lds_b128((gv4i_p)g, (lv4i_p)lds, 0, 0);
#else
  *(v8h*)lds = *(const v8h*)g;
#endif
}

__device__ __forceinline__ void wait_async() {
#if HAS_ASYNC_LDS
#if __has_builtin(__builtin_amdgcn_s_wait_asynccnt)
  __builtin_amdgcn_s_wait_asynccnt(0);
#else
  asm volatile("s_wait_asynccnt 0" ::: "memory");
#endif
#endif
}

// ---------------------------------------------------------------------------
// WMMA 16x16x32 f16 fragment loader.
// A-matrix (16 rows x 32 K, row-major, leading dim `ld` in halves):
//   lanes 0-15  : row = lane,     elems 0-7 = K 0..7,   elems 8-15 = K 16..23
//   lanes 16-31 : row = lane-16,  elems 0-7 = K 8..15,  elems 8-15 = K 24..31
// B-matrix (32 K x 16 cols) uses the symmetric layout: lane = column, elems
// run along K (source stores each column contiguously along K).
// Requires base 16B-aligned and ld % 8 == 0.
// ---------------------------------------------------------------------------
__device__ __forceinline__ v16h load_frag(const _Float16* base, int ld) {
  const int lane = threadIdx.x & 31;
  const _Float16* p = base + (lane & 15) * ld + (lane >> 4) * 8;
  v8h lo = *(const v8h*)(p);
  v8h hi = *(const v8h*)(p + 16);
  v16h f;
#pragma unroll
  for (int i = 0; i < 8; ++i) { f[i] = lo[i]; f[i + 8] = hi[i]; }
  return f;
}

// ---------------------------------------------------------------------------
// f32 -> f16 strided conversion (obs + weight prep)
// ---------------------------------------------------------------------------
__global__ void cvt_f32_f16(const float* __restrict__ src, _Float16* __restrict__ dst,
                            int rows, int cols, int sld, int dld) {
  int i = blockIdx.x * blockDim.x + threadIdx.x;
  if (i >= rows * cols) return;
  int r = i / cols, c = i - r * cols;
  dst[(size_t)r * dld + c] = (_Float16)src[(size_t)r * sld + c];
}

// ---------------------------------------------------------------------------
// Pre-transpose V so attention B-fragments are contiguous 16B loads:
// vT[((b*H + h)*DH + d)*N + n] = qkv[(b*N + n)*3E + 2E + h*DH + d]
// ---------------------------------------------------------------------------
__global__ void transpose_v(const _Float16* __restrict__ qkvb, _Float16* __restrict__ vT) {
  int i = blockIdx.x * blockDim.x + threadIdx.x;
  if (i >= BB * HH * DHH * NN) return;
  int n = i & (NN - 1);
  int d = (i >> 8) & (DHH - 1);
  int h = (i >> 13) & (HH - 1);
  int b = i >> 16;
  vT[i] = qkvb[((size_t)(b * NN + n)) * (3 * EE) + 2 * EE + h * DHH + d];
}

// ---------------------------------------------------------------------------
// Generic f16 GEMM with fused epilogues:   D = epilogue(A @ Bw^T)
//   A   : (MTOT x Ktot) f16, row-major, lda
//   Bw  : (Ntot x Ktot) f16 weight rows, row-major, ldb   (computes A @ Bw^T)
// Block = 256 threads (8 waves), tile 128(M) x 64(N), K-step 32.
// Wave (wm,wn) in 4x2 owns a 32x32 micro-tile = 2x2 WMMA accumulators.
// Tiles are staged via GLOBAL_LOAD_ASYNC_TO_LDS_B128 (ASYNCcnt); next K-tile
// is prefetched (global_prefetch_b8).
// MODE 1: + b + ratio*W_in[:,D] then ReLU   (input projection)
// MODE 2: + b                               (qkv projection)
// MODE 3: + b + residual(h)                 (attention out-projection)
// MODE 4: + b then ReLU                     (output projection)
// ---------------------------------------------------------------------------
template <int MODE>
__global__ void gemm_f16(const _Float16* __restrict__ A, int lda,
                         const _Float16* __restrict__ Bw, int ldb,
                         _Float16* __restrict__ Dst, int ldd, int Ktot,
                         const float* __restrict__ bias,
                         const float* __restrict__ extra,    // MODE 1: W_in (f32, ld 257)
                         const _Float16* __restrict__ resid, // MODE 3: h
                         const int* __restrict__ actv) {
  __shared__ _Float16 As[128 * 40];
  __shared__ _Float16 Bs[64 * 40];

  const int tid   = threadIdx.x;
  const int lane  = tid & 31;
  const int wv    = tid >> 5;
  const int waveM = wv & 3;
  const int waveN = wv >> 2;
  const int m0    = blockIdx.y * 128;
  const int n0    = blockIdx.x * 64;

  // per-thread staging coordinates
  const int ar0 = tid >> 2,          akc0 = (tid & 3) << 3;          // A chunk 0
  const int ar1 = (tid + 256) >> 2,  akc1 = akc0;                    // A chunk 1
  const int br  = tid >> 2,          bkc  = (tid & 3) << 3;          // B chunk

  v8f acc[2][2] = {};

  for (int k0 = 0; k0 < Ktot; k0 += 32) {
    __syncthreads();
    // A tile: 128 rows x 32 halves, async 16B/lane straight into LDS
    async_copy16(&As[ar0 * 40 + akc0], &A[(size_t)(m0 + ar0) * lda + k0 + akc0]);
    async_copy16(&As[ar1 * 40 + akc1], &A[(size_t)(m0 + ar1) * lda + k0 + akc1]);
    // B tile: 64 weight rows x 32 halves
    async_copy16(&Bs[br * 40 + bkc], &Bw[(size_t)(n0 + br) * ldb + k0 + bkc]);

    // prefetch next K-tile while this one lands
    if (k0 + 32 < Ktot) {
      __builtin_prefetch(&A[(size_t)(m0 + ar0) * lda + k0 + 32 + akc0], 0, 1);
      __builtin_prefetch(&Bw[(size_t)(n0 + br) * ldb + k0 + 32 + bkc], 0, 1);
    }

    wait_async();
    __syncthreads();

    v16h af[2], bf[2];
#pragma unroll
    for (int mi = 0; mi < 2; ++mi)
      af[mi] = load_frag(&As[(waveM * 32 + mi * 16) * 40], 40);
#pragma unroll
    for (int ni = 0; ni < 2; ++ni)
      bf[ni] = load_frag(&Bs[(waveN * 32 + ni * 16) * 40], 40);

#pragma unroll
    for (int mi = 0; mi < 2; ++mi)
#pragma unroll
      for (int ni = 0; ni < 2; ++ni)
        acc[mi][ni] = __builtin_amdgcn_wmma_f32_16x16x32_f16(
            false, af[mi], false, bf[ni], (short)0, acc[mi][ni], false, false);
  }

  // Epilogue. C/D layout: lanes 0-15 -> M = j, col = lane; lanes 16-31 -> M = 8+j.
  const int colb = lane & 15;
  const int rowq = (lane >> 4) * 8;
#pragma unroll
  for (int mi = 0; mi < 2; ++mi)
#pragma unroll
    for (int ni = 0; ni < 2; ++ni) {
      const int ncol = n0 + waveN * 32 + ni * 16 + colb;
      const int mrow = m0 + waveM * 32 + mi * 16 + rowq;
#pragma unroll
      for (int j = 0; j < 8; ++j) {
        const int row = mrow + j;
        float v = acc[mi][ni][j] + bias[ncol];
        if (MODE == 1) {
          const float ratio = (float)actv[row >> 8] * (1.0f / (float)NN);
          v += ratio * extra[ncol * (DD + 1) + DD];  // rank-1 term for x[:, D]
          v = fmaxf(v, 0.0f);
        }
        if (MODE == 3) v += (float)resid[(size_t)row * ldd + ncol];
        if (MODE == 4) v = fmaxf(v, 0.0f);
        Dst[(size_t)row * ldd + ncol] = (_Float16)v;
      }
    }
}

// ---------------------------------------------------------------------------
// Fused masked attention. One wave per (batch, head, 16-query tile).
// Block = 64 threads = 2 waves. Scores in f32 LDS (16x256 per wave), softmax
// with key-mask, probs in f16 LDS, ctx = P @ V via WMMA over K = 256 keys.
// ---------------------------------------------------------------------------
__global__ void attn_kernel(const _Float16* __restrict__ qkvb,
                            const _Float16* __restrict__ vT,
                            const int* __restrict__ actv,
                            _Float16* __restrict__ ctxb) {
  __shared__ float    sc[2][16][NN];
  __shared__ _Float16 pr[2][16][NN];

  const int wave = threadIdx.x >> 5;
  const int lane = threadIdx.x & 31;
  const int tile = blockIdx.x * 2 + wave;     // 0 .. B*H*(N/16)-1
  const int qt   = tile & 15;
  const int head = (tile >> 4) & (HH - 1);
  const int bt   = tile >> 7;
  const int act  = actv[bt];
  const int rowBase = bt * NN + qt * 16;

  const int colb = lane & 15;
  const int rowq = (lane >> 4) * 8;
  const float scale = 0.17677669529663687f;   // 1/sqrt(DH)

  // Q fragment: 16 queries x 32 dims
  v16h qf = load_frag(qkvb + (size_t)rowBase * (3 * EE) + head * DHH, 3 * EE);

  // ---- scores: one 16x16x32 WMMA per key tile ----
  for (int kt = 0; kt < NN / 16; ++kt) {
    const int keyBase = bt * NN + kt * 16;
    v16h kf = load_frag(qkvb + (size_t)keyBase * (3 * EE) + EE + head * DHH, 3 * EE);
    v8f z = {};
    v8f s = __builtin_amdgcn_wmma_f32_16x16x32_f16(false, qf, false, kf, (short)0, z,
                                                   false, false);
    const int key = kt * 16 + colb;
#pragma unroll
    for (int j = 0; j < 8; ++j)
      sc[wave][rowq + j][key] = (key < act) ? s[j] * scale : -1.0e9f;
  }

  // ---- softmax (lanes 0-15 each own one query row) ----
  if (lane < 16) {
    float m = -3.4e38f;
    for (int c = 0; c < NN; ++c) m = fmaxf(m, sc[wave][lane][c]);
    float sum = 0.0f;
    for (int c = 0; c < NN; ++c) {
      float e = expf(sc[wave][lane][c] - m);
      sum += e;
      sc[wave][lane][c] = e;
    }
    const float inv = 1.0f / sum;
    for (int c = 0; c < NN; ++c)
      pr[wave][lane][c] = (_Float16)(sc[wave][lane][c] * inv);
  }

  // ---- ctx = P(16x256) @ V(256x32): 8 K-chunks x 2 N-halves of WMMA ----
  v8f acc[2] = {};
  const _Float16* vbase = vT + (size_t)(bt * HH + head) * DHH * NN;
  for (int c = 0; c < NN / 32; ++c) {
    v16h pf = load_frag(&pr[wave][0][0] + c * 32, NN);
#pragma unroll
    for (int ni = 0; ni < 2; ++ni) {
      v16h vf = load_frag(vbase + (size_t)(ni * 16) * NN + c * 32, NN);
      acc[ni] = __builtin_amdgcn_wmma_f32_16x16x32_f16(false, pf, false, vf, (short)0,
                                                       acc[ni], false, false);
    }
  }
#pragma unroll
  for (int ni = 0; ni < 2; ++ni)
#pragma unroll
    for (int j = 0; j < 8; ++j) {
      const int m = rowq + j;
      const int d = ni * 16 + colb;
      ctxb[(size_t)(rowBase + m) * EE + head * DHH + d] = (_Float16)acc[ni][j];
    }
}

// ---------------------------------------------------------------------------
// Value head: val = leaky_relu(rsa @ W_v^T + b_v), masked sum over agents.
// One block per batch; writes out[b] exactly once (graph-capture safe).
// ---------------------------------------------------------------------------
__global__ void value_kernel(const _Float16* __restrict__ rsa,
                             const float* __restrict__ W_v,
                             const float* __restrict__ b_v,
                             const int* __restrict__ actv,
                             float* __restrict__ out) {
  __shared__ float red[NN];
  const int b = blockIdx.x;
  const int n = threadIdx.x;
  const int act = actv[b];
  const _Float16* row = rsa + (size_t)(b * NN + n) * EE;
  float dot = 0.0f;
  for (int e = 0; e < EE; ++e) dot += (float)row[e] * W_v[e];
  float val = dot + b_v[0];
  val = (val > 0.0f) ? val : 0.01f * val;   // leaky_relu, slope 0.01
  red[n] = (n < act) ? val : 0.0f;
  __syncthreads();
  for (int s = NN / 2; s > 0; s >>= 1) {
    if (n < s) red[n] += red[n + s];
    __syncthreads();
  }
  if (n == 0) out[b] = red[0];
}

// ---------------------------------------------------------------------------
extern "C" void kernel_launch(void* const* d_in, const int* in_sizes, int n_in,
                              void* d_out, int out_size, void* d_ws, size_t ws_size,
                              hipStream_t stream) {
  (void)in_sizes; (void)n_in; (void)out_size; (void)ws_size;

  const float* obs   = (const float*)d_in[0];
  const int*   actv  = (const int*)d_in[1];
  const float* W_in  = (const float*)d_in[2];
  const float* b_in  = (const float*)d_in[3];
  const float* W_qkv = (const float*)d_in[4];
  const float* b_qkv = (const float*)d_in[5];
  const float* W_o   = (const float*)d_in[6];
  const float* b_o   = (const float*)d_in[7];
  const float* W_out = (const float*)d_in[8];
  const float* b_out = (const float*)d_in[9];
  const float* W_v   = (const float*)d_in[10];
  const float* b_v   = (const float*)d_in[11];
  float* out = (float*)d_out;

  // ---- workspace layout (halves) ----
  _Float16* ws     = (_Float16*)d_ws;
  _Float16* d_x    = ws;                                 // 16384*256
  _Float16* d_h    = d_x    + (size_t)MTOT * EE;
  _Float16* d_qkv  = d_h    + (size_t)MTOT * EE;         // 16384*768
  _Float16* d_ctx  = d_qkv  + (size_t)MTOT * 3 * EE;
  _Float16* d_t    = d_ctx  + (size_t)MTOT * EE;
  _Float16* d_rsa  = d_t    + (size_t)MTOT * EE;
  _Float16* d_vT   = d_rsa  + (size_t)MTOT * EE;         // B*H*DH*N
  _Float16* w_in   = d_vT   + (size_t)MTOT * EE;
  _Float16* w_qkv  = w_in   + (size_t)EE * DD;
  _Float16* w_o    = w_qkv  + (size_t)3 * EE * EE;
  _Float16* w_out  = w_o    + (size_t)EE * EE;

  // ---- f32 -> f16 prep ----
  cvt_f32_f16<<<(MTOT * EE + 255) / 256, 256, 0, stream>>>(obs, d_x, MTOT, EE, DD, EE);
  cvt_f32_f16<<<(EE * DD + 255) / 256, 256, 0, stream>>>(W_in, w_in, EE, DD, DD + 1, DD);
  cvt_f32_f16<<<(3 * EE * EE + 255) / 256, 256, 0, stream>>>(W_qkv, w_qkv, 3 * EE, EE, EE, EE);
  cvt_f32_f16<<<(EE * EE + 255) / 256, 256, 0, stream>>>(W_o, w_o, EE, EE, EE, EE);
  cvt_f32_f16<<<(EE * EE + 255) / 256, 256, 0, stream>>>(W_out, w_out, EE, EE, EE, EE);

  // ---- h = relu(x @ W_in^T + b_in + ratio * W_in[:,D]) ----
  gemm_f16<1><<<dim3(EE / 64, MTOT / 128), 256, 0, stream>>>(
      d_x, EE, w_in, DD, d_h, EE, DD, b_in, W_in, nullptr, actv);

  // ---- qkv = h @ W_qkv^T + b_qkv ----
  gemm_f16<2><<<dim3(3 * EE / 64, MTOT / 128), 256, 0, stream>>>(
      d_h, EE, w_qkv, EE, d_qkv, 3 * EE, EE, b_qkv, nullptr, nullptr, nullptr);

  // ---- V transpose for attention B-operand ----
  transpose_v<<<(BB * HH * DHH * NN + 255) / 256, 256, 0, stream>>>(d_qkv, d_vT);

  // ---- fused masked self-attention ----
  attn_kernel<<<(BB * HH * (NN / 16)) / 2, 64, 0, stream>>>(d_qkv, d_vT, actv, d_ctx);

  // ---- t = ctx @ W_o^T + b_o + h ----
  gemm_f16<3><<<dim3(EE / 64, MTOT / 128), 256, 0, stream>>>(
      d_ctx, EE, w_o, EE, d_t, EE, EE, b_o, nullptr, d_h, nullptr);

  // ---- rsa = relu(t @ W_out^T + b_out) ----
  gemm_f16<4><<<dim3(EE / 64, MTOT / 128), 256, 0, stream>>>(
      d_t, EE, w_out, EE, d_rsa, EE, EE, b_out, nullptr, nullptr, nullptr);

  // ---- masked value sum ----
  value_kernel<<<BB, NN, 0, stream>>>(d_rsa, W_v, b_v, actv, out);
}
